// MessageFunction_8796093022562
// MI455X (gfx1250) — matrix operand order repaired and checked by reading, not compiled, not run on hardware
//
#include <hip/hip_runtime.h>

typedef __attribute__((ext_vector_type(16))) _Float16 v16h;
typedef __attribute__((ext_vector_type(8)))  _Float16 v8h;
typedef __attribute__((ext_vector_type(8)))  float    v8f;
typedef __attribute__((ext_vector_type(4)))  float    v4f;

#define WAVES_PER_BLOCK 8

// ---------------------------------------------------------------------------
// Prep: transpose + fp32->fp16 convert W1 (16x64) -> W1T (64x16) and
// W2 (64x256) -> W2T (256x64); copy b2 (256 f32). All into d_ws.
// ---------------------------------------------------------------------------
__global__ void prep_kernel(const float* __restrict__ W1,
                            const float* __restrict__ W2,
                            const float* __restrict__ b2,
                            _Float16* __restrict__ w2t,
                            _Float16* __restrict__ w1t,
                            float* __restrict__ b2c) {
  int tid = threadIdx.x;
  for (int i = tid; i < 256 * 64; i += 256) {
    int col = i >> 6, k = i & 63;           // w2t[col][k] = W2[k][col]
    w2t[i] = (_Float16)W2[k * 256 + col];
  }
  for (int i = tid; i < 64 * 16; i += 256) {
    int col = i >> 4, k = i & 15;           // w1t[col][k] = W1[k][col]
    w1t[i] = (_Float16)W1[k * 64 + col];
  }
  if (tid < 256) b2c[tid] = b2[tid];
}

// ---------------------------------------------------------------------------
// Fused edge-MLP + per-edge matvec + scatter-add. One wave handles 16 edges.
// ---------------------------------------------------------------------------
__launch_bounds__(256)
__global__ void msg_kernel(const int*   __restrict__ idxv,
                           const float* __restrict__ h_w,
                           const float* __restrict__ e_vw,
                           const float* __restrict__ b1,
                           const _Float16* __restrict__ w2t_g,
                           const _Float16* __restrict__ w1t_g,
                           const float* __restrict__ b2_g,
                           float* __restrict__ out,
                           int n_edge, int n_tiles) {
  __shared__ _Float16 sW2T[256 * 64];                 // 32 KB, row-major [col][k]
  __shared__ _Float16 sW1T[64 * 16];                  //  2 KB, row-major [col][k]
  __shared__ float    sB2[256];                       //  1 KB
  __shared__ _Float16 sHid[WAVES_PER_BLOCK][16 * 64]; // 16 KB, per-wave hid tile

  const int tid = threadIdx.x;

  // Coalesced stage of preprocessed weights into LDS.
  {
    const uint4* s2 = (const uint4*)w2t_g;
    uint4*       d2 = (uint4*)sW2T;
    for (int i = tid; i < (256 * 64 * 2) / 16; i += 256) d2[i] = s2[i];
    const uint4* s1 = (const uint4*)w1t_g;
    uint4*       d1 = (uint4*)sW1T;
    for (int i = tid; i < (64 * 16 * 2) / 16; i += 256) d1[i] = s1[i];
    if (tid < 256) sB2[tid] = b2_g[tid];
  }
  __syncthreads();

  const int lane = tid & 31;
  const int wave = tid >> 5;
  const int lq   = lane & 15;   // edge-in-tile / row index
  const int hh   = lane >> 4;   // lane half

  // b1 broadcast values for GEMM1 C-init (C layout: N = lane%16).
  float b1v[4];
#pragma unroll
  for (int j = 0; j < 4; ++j) b1v[j] = b1[16 * j + lq];

  // W1 B-operands (K=8*hh+s in elems 0-7 of each half, elems 8-15 zero = K 16..31).
  v16h w1b[4];
#pragma unroll
  for (int j = 0; j < 4; ++j) {
    v16h t = {};
    v8h u = *(const v8h*)&sW1T[(16 * j + lq) * 16 + 8 * hh];
#pragma unroll
    for (int i = 0; i < 8; ++i) t[i] = u[i];
    w1b[j] = t;
  }

  _Float16* hid = &sHid[wave][0];

  const int gw = blockIdx.x * WAVES_PER_BLOCK + wave;
  const int nw = gridDim.x * WAVES_PER_BLOCK;

  for (int tile = gw; tile < n_tiles; tile += nw) {
    const int  edge = tile * 16 + lq;
    const bool ok   = edge < n_edge;
    const int  ec   = ok ? edge : (n_edge - 1);

    // ---- A-operand from e_vw: lane row = lq, K 0..7 (lo half) / 8..15 (hi).
    const float* ep = e_vw + (size_t)ec * 16 + 8 * hh;
    v4f f0 = *(const v4f*)ep;
    v4f f1 = *(const v4f*)(ep + 4);
    v16h a1 = {};
    a1[0] = (_Float16)f0[0]; a1[1] = (_Float16)f0[1];
    a1[2] = (_Float16)f0[2]; a1[3] = (_Float16)f0[3];
    a1[4] = (_Float16)f1[0]; a1[5] = (_Float16)f1[1];
    a1[6] = (_Float16)f1[2]; a1[7] = (_Float16)f1[3];   // elems 8..15 stay 0 (K pad)

    // ---- h in f32, matching D layout: hreg[r] = h[lq][r + 8*hh].
    const float* hp = h_w + (size_t)ec * 16 + 8 * hh;
    v4f h0 = *(const v4f*)hp;
    v4f h1 = *(const v4f*)(hp + 4);
    float hreg[8] = {h0[0], h0[1], h0[2], h0[3], h1[0], h1[1], h1[2], h1[3]};

    const int idx = idxv[ec];
    float* obase = out + (size_t)idx * 16 + 8 * hh;

    // ---- GEMM1: hid(16x64) = relu(e @ W1 + b1), written f16 row-major to LDS.
#pragma unroll
    for (int j = 0; j < 4; ++j) {
      v8f c;
#pragma unroll
      for (int r = 0; r < 8; ++r) c[r] = b1v[j];
      v8f d = __builtin_amdgcn_wmma_f32_16x16x32_f16(false, a1, false, w1b[j],
                                                     (short)0, c, false, false);
#pragma unroll
      for (int r = 0; r < 8; ++r) {
        float v = __builtin_fmaxf(d[r], 0.0f);               // single v_max_num_f32
        hid[(r + 8 * hh) * 64 + 16 * j + lq] = (_Float16)v;  // [edge][k]
      }
    }
    __asm volatile("" ::: "memory");  // keep LDS store->load order (DS in-order)

    // ---- hid as B-operand (N = edge across lanes, K across VGPRs/halves).
    v16h hB0 = *(const v16h*)&hid[lq * 64 + 0  + 16 * hh];
    v16h hB1 = *(const v16h*)&hid[lq * 64 + 32 + 16 * hh];

    // ---- GEMM2 transposed + fused matvec + scatter.
    // t indexes output column m; pair (p, p+8) so both halves do atomics.
#pragma unroll
    for (int p = 0; p < 8; ++p) {
      float s[2];
#pragma unroll
      for (int q = 0; q < 2; ++q) {
        const int t = p + 8 * q;
        // C-init from b2: C[r] = b2[16t + r + 8*hh]  (n = r + 8*hh).
        const float* bp = &sB2[16 * t + 8 * hh];
        v4f c0 = *(const v4f*)bp;
        v4f c1 = *(const v4f*)(bp + 4);
        v8f c = {c0[0], c0[1], c0[2], c0[3], c1[0], c1[1], c1[2], c1[3]};
        // A-operands: rows = W2T[16t + lq][.], chunk k = 0..31 and 32..63.
        const _Float16* wrow = &sW2T[(16 * t + lq) * 64];
        v8h a0lo = *(const v8h*)&wrow[0  + 16 * hh];
        v8h a0hi = *(const v8h*)&wrow[8  + 16 * hh];
        v8h b0lo = *(const v8h*)&wrow[32 + 16 * hh];
        v8h b0hi = *(const v8h*)&wrow[40 + 16 * hh];
        v16h A0, A1;
#pragma unroll
        for (int i = 0; i < 8; ++i) {
          A0[i] = a0lo[i]; A0[i + 8] = a0hi[i];
          A1[i] = b0lo[i]; A1[i + 8] = b0hi[i];
        }
        v8f d = __builtin_amdgcn_wmma_f32_16x16x32_f16(false, A0, false, hB0,
                                                       (short)0, c, false, false);
        d = __builtin_amdgcn_wmma_f32_16x16x32_f16(false, A1, false, hB1,
                                                   (short)0, d, false, false);
        // D[r][lane] = A_vw[edge=lq][m=t][n=r+8*hh]; dot with h over this half.
        float acc = d[0] * hreg[0];
#pragma unroll
        for (int r = 1; r < 8; ++r) acc = fmaf(d[r], hreg[r], acc);
        s[q] = acc;
      }
      // Combine halves: lanes 0-15 -> m=p, lanes 16-31 -> m=p+8.
      float z = hh ? s[1] : s[0];
      float w = hh ? s[0] : s[1];
      w = __shfl_xor(w, 16, 32);
      float res = z + w;
      if (ok) unsafeAtomicAdd(obase + p, res);  // out[idx*16 + p + 8*hh]
    }
  }
}

// ---------------------------------------------------------------------------
extern "C" void kernel_launch(void* const* d_in, const int* in_sizes, int n_in,
                              void* d_out, int out_size, void* d_ws, size_t ws_size,
                              hipStream_t stream) {
  const int*   index_v = (const int*)d_in[0];
  const float* h_w     = (const float*)d_in[1];
  const float* e_vw    = (const float*)d_in[2];
  // d_in[3] = n_node (scalar, unused; out_size gives output extent)
  const float* W1 = (const float*)d_in[4];
  const float* b1 = (const float*)d_in[5];
  const float* W2 = (const float*)d_in[6];
  const float* b2 = (const float*)d_in[7];

  const int n_edge  = in_sizes[0];
  const int n_tiles = (n_edge + 15) / 16;

  _Float16* w2t = (_Float16*)d_ws;                    // 32768 B
  _Float16* w1t = (_Float16*)((char*)d_ws + 32768);   //  2048 B
  float*    b2c = (float*)((char*)d_ws + 34816);      //  1024 B

  hipMemsetAsync(d_out, 0, (size_t)out_size * sizeof(float), stream);
  prep_kernel<<<1, 256, 0, stream>>>(W1, W2, b2, w2t, w1t, b2c);

  int blocks = (n_tiles + WAVES_PER_BLOCK - 1) / WAVES_PER_BLOCK;
  if (blocks > 1024) blocks = 1024;
  if (blocks < 1)    blocks = 1;
  msg_kernel<<<blocks, 256, 0, stream>>>(index_v, h_w, e_vw, b1, w2t, w1t, b2c,
                                         (float*)d_out, n_edge, n_tiles);
}
